// BaselineGPT_72524817760542
// MI455X (gfx1250) — matrix operand order repaired and checked by reading, not compiled, not run on hardware
//
#include <hip/hip_runtime.h>
#include <hip/hip_bf16.h>

#define BB 2
#define SS 2048
#define DMODEL 1024
#define NH 16
#define NKV 4
#define HDIM 64
#define WIN 256

typedef __attribute__((ext_vector_type(16))) _Float16 v16h;
typedef __attribute__((ext_vector_type(8)))  float    v8f;

__device__ __forceinline__ v8f wmma16(v16h a, v16h b, v8f c) {
  return __builtin_amdgcn_wmma_f32_16x16x32_f16(false, a, false, b, (short)0, c,
                                                false, false);
}
// A-fragment (16x32 f16): lane r=l&15 is row, half kh=l>>4; 16 contiguous
// halves at K = kh*16 (matches ldB's K order).
__device__ __forceinline__ v16h ldA(const _Float16* p) {
  v16h f;
  ((uint4*)&f)[0] = *(const uint4*)(p);
  ((uint4*)&f)[1] = *(const uint4*)(p + 8);
  return f;
}
// B-fragment (32x16 f16): lane n=l&15 is column, half selects K range
// [kh*16, kh*16+16) -> 16 contiguous halves (this code's order, same as ldA).
__device__ __forceinline__ v16h ldB(const _Float16* p) {
  v16h f;
  ((uint4*)&f)[0] = ((const uint4*)p)[0];
  ((uint4*)&f)[1] = ((const uint4*)p)[1];
  return f;
}

// Async DMA: 16 bytes per lane, global -> LDS, tracked by ASYNCcnt.
// lds_off = LDS byte offset (low 32 bits of the generic shared address).
__device__ __forceinline__ void async_cp_b128(unsigned lds_off,
                                              const _Float16* gptr) {
  asm volatile("global_load_async_to_lds_b128 %0, %1, off"
               :
               : "v"(lds_off), "v"(gptr)
               : "memory");
}
__device__ __forceinline__ unsigned lds_off32(const void* p) {
  return (unsigned)(size_t)p;  // LDS aperture keeps the offset in addr[31:0]
}

__global__ void cvt_f16_kernel(const float* __restrict__ src,
                               _Float16* __restrict__ dst, int n) {
  int i = blockIdx.x * blockDim.x + threadIdx.x;
  int stride = gridDim.x * blockDim.x;
  for (; i < n; i += stride) dst[i] = (_Float16)src[i];
}

// C[m,n] = sum_k A[m,k] * B[n,k]; A: MxK row-major f16, B: NxK row-major f16.
// Block tile 128x64, BK=32, 8 waves each computing a 32x32 subtile.
// Double-buffered LDS fed by global_load_async_to_lds_b128.
__global__ __launch_bounds__(256) void gemm_nt_wmma(
    const _Float16* __restrict__ A, const _Float16* __restrict__ Bw,
    float* __restrict__ Cf, _Float16* __restrict__ Ch, int M, int N, int K) {
  __shared__ __align__(16) _Float16 As[2][128 * 40];
  __shared__ __align__(16) _Float16 Bs[2][64 * 40];
  const int tid = threadIdx.x;
  const int lane = tid & 31;
  const int wid = tid >> 5;
  const int wm = wid >> 1, wn = wid & 1;
  const int r = lane & 15, khf = lane >> 4;
  const int bm = blockIdx.y, bn = blockIdx.x;

  // Per-thread copy slots: A tile 128x32 halves = 512 x 16B (2 per thread),
  // B tile 64x32 halves = 256 x 16B (1 per thread).
  const int ar0 = tid >> 2, ac = (tid & 3) * 8;  // rows 0..63
  const int ar1 = ar0 + 64;                      // rows 64..127
  const int br = tid >> 2, bc = (tid & 3) * 8;
  const _Float16* gA0 = &A[(size_t)(bm * 128 + ar0) * K + ac];
  const _Float16* gA1 = &A[(size_t)(bm * 128 + ar1) * K + ac];
  const _Float16* gB = &Bw[(size_t)(bn * 64 + br) * K + bc];
  unsigned lA0[2], lA1[2], lB[2];
#pragma unroll
  for (int bf = 0; bf < 2; ++bf) {
    lA0[bf] = lds_off32(&As[bf][ar0 * 40 + ac]);
    lA1[bf] = lds_off32(&As[bf][ar1 * 40 + ac]);
    lB[bf] = lds_off32(&Bs[bf][br * 40 + bc]);
  }

  auto issue = [&](int bf, int k0) {
    async_cp_b128(lA0[bf], gA0 + k0);
    async_cp_b128(lA1[bf], gA1 + k0);
    async_cp_b128(lB[bf], gB + k0);
  };

  v8f acc00 = {}, acc01 = {}, acc10 = {}, acc11 = {};

  issue(0, 0);
  if (K > 32) issue(1, 32);
  asm volatile("s_wait_asynccnt 0x3" ::: "memory");  // buffer 0 landed
  __syncthreads();

  int cur = 0;
  for (int k0 = 0; k0 < K; k0 += 32) {
    const _Float16* pa = &As[cur][(wm * 32 + r) * 40 + khf * 16];
    v16h a0 = ldA(pa);
    v16h a1 = ldA(pa + 16 * 40);
    const _Float16* pb = &Bs[cur][(wn * 32 + r) * 40 + khf * 16];
    v16h b0 = ldB(pb);
    v16h b1 = ldB(pb + 16 * 40);
    acc00 = wmma16(a0, b0, acc00);
    acc01 = wmma16(a0, b1, acc01);
    acc10 = wmma16(a1, b0, acc10);
    acc11 = wmma16(a1, b1, acc11);

    __syncthreads();  // all waves done reading buffer `cur`
    if (k0 + 64 < K) {
      issue(cur, k0 + 64);  // refill the buffer we just finished reading
      if (k0 + 96 < K) {
        __builtin_prefetch(gA0 + k0 + 96, 0, 1);
        __builtin_prefetch(gB + k0 + 96, 0, 1);
      }
      // 6 outstanding -> wait until only the newest batch remains:
      asm volatile("s_wait_asynccnt 0x3" ::: "memory");
    } else {
      asm volatile("s_wait_asynccnt 0x0" ::: "memory");  // drain
    }
    __syncthreads();  // next buffer is now visible to every wave
    cur ^= 1;
  }

  const int rb = bm * 128 + wm * 32;
  const int cb = bn * 64 + wn * 32;
#pragma unroll
  for (int i = 0; i < 8; ++i) {
    int row0 = rb + i + 8 * khf;
    int row1 = row0 + 16;
    int col0 = cb + r;
    int col1 = col0 + 16;
    if (Cf) {
      Cf[(size_t)row0 * N + col0] = acc00[i];
      Cf[(size_t)row0 * N + col1] = acc01[i];
      Cf[(size_t)row1 * N + col0] = acc10[i];
      Cf[(size_t)row1 * N + col1] = acc11[i];
    } else {
      Ch[(size_t)row0 * N + col0] = (_Float16)acc00[i];
      Ch[(size_t)row0 * N + col1] = (_Float16)acc01[i];
      Ch[(size_t)row1 * N + col0] = (_Float16)acc10[i];
      Ch[(size_t)row1 * N + col1] = (_Float16)acc11[i];
    }
  }
}

// Per (b, head, s): RMSNorm over 64, RoPE, optional gain; wave-per-row.
__global__ void rms_rope_kernel(const _Float16* __restrict__ lin,
                                _Float16* __restrict__ out,
                                const float* __restrict__ gain, int nh) {
  int gw = (blockIdx.x * blockDim.x + threadIdx.x) >> 5;
  int lane = threadIdx.x & 31;
  if (gw >= BB * nh * SS) return;
  int s = gw % SS;
  int t = gw / SS;
  int h = t % nh;
  int b = t / nh;
  const _Float16* src = lin + ((size_t)(b * SS + s)) * (nh * HDIM) + h * HDIM;
  float v1 = (float)src[lane];
  float v2 = (float)src[lane + 32];
  float ss = v1 * v1 + v2 * v2;
#pragma unroll
  for (int xm = 1; xm < 32; xm <<= 1) ss += __shfl_xor(ss, xm, 32);
  float rn = rsqrtf(ss * (1.0f / 64.0f) + 1e-6f);
  float invf = __expf(-9.210340372f * ((float)lane * (1.0f / 32.0f)));
  float sn, cs;
  __sincosf((float)s * invf, &sn, &cs);
  float g = gain ? gain[h] : 1.0f;
  float q1 = v1 * rn, q2 = v2 * rn;
  _Float16* dst = out + (((size_t)(b * nh + h)) * SS + s) * HDIM;
  dst[lane] = (_Float16)((q1 * cs + q2 * sn) * g);
  dst[lane + 32] = (_Float16)((-q1 * sn + q2 * cs) * g);
}

// v_lin [b*S+s][g*64+d] -> vt [b][g][d][s]
__global__ void v_transpose_kernel(const _Float16* __restrict__ vlin,
                                   _Float16* __restrict__ vt) {
  int n = BB * NKV * HDIM * SS;
  for (int o = blockIdx.x * blockDim.x + threadIdx.x; o < n;
       o += gridDim.x * blockDim.x) {
    int s = o & (SS - 1);
    int d = (o >> 11) & (HDIM - 1);
    int g = (o >> 17) & (NKV - 1);
    int b = o >> 19;
    vt[o] = vlin[((size_t)(b * SS + s)) * (NKV * HDIM) + g * HDIM + d];
  }
}

// Flash attention, one wave per (b, h, 16-query block), 32-key chunks.
__global__ __launch_bounds__(128) void attn_kernel(
    const _Float16* __restrict__ qh, const _Float16* __restrict__ kh,
    const _Float16* __restrict__ vt, _Float16* __restrict__ y) {
  __shared__ __align__(16) _Float16 att_s[4][16 * 40];
  const int lane = threadIdx.x & 31;
  const int wid = threadIdx.x >> 5;
  const int wg = blockIdx.x * 4 + wid;
  const int qb = wg & 127;
  const int h = (wg >> 7) & 15;
  const int b = wg >> 11;
  const int q0 = qb * 16;
  const int r = lane & 15, khalf = lane >> 4;

  const _Float16* qbase = qh + ((size_t)(b * NH + h)) * SS * HDIM;
  const _Float16* kbase = kh + ((size_t)(b * NKV + (h >> 2))) * SS * HDIM;
  const _Float16* vbase = vt + ((size_t)(b * NKV + (h >> 2))) * HDIM * SS;
  _Float16* amy = att_s[wid];

  v16h qa0 = ldA(qbase + (size_t)(q0 + r) * HDIM + khalf * 16);
  v16h qa1 = ldA(qbase + (size_t)(q0 + r) * HDIM + 32 + khalf * 16);

  v8f acc0 = {}, acc1 = {}, acc2 = {}, acc3 = {};
  float rmax[8], rsum[8];
#pragma unroll
  for (int i = 0; i < 8; ++i) { rmax[i] = -3.0e38f; rsum[i] = 0.0f; }

  int klo = q0 - (WIN - 1);
  if (klo < 0) klo = 0;
  const int kc0 = klo & ~31;
  const float scale = 0.125f;  // 1/sqrt(64)

  for (int kc = kc0; kc <= q0 + 15; kc += 32) {
    float sc[2][8];
#pragma unroll
    for (int j = 0; j < 2; ++j) {
      int kb = kc + 16 * j;
      int key = kb + r;                    // key index for C columns
      int keyc = key < SS ? key : SS - 1;  // clamp loads in-bounds
      v16h kf0 = ldB(kbase + (size_t)keyc * HDIM + khalf * 16);
      v16h kf1 = ldB(kbase + (size_t)keyc * HDIM + 32 + khalf * 16);
      v8f s = {};
      s = wmma16(qa0, kf0, s);
      s = wmma16(qa1, kf1, s);
#pragma unroll
      for (int i = 0; i < 8; ++i) {
        int m = i + 8 * khalf;
        int qpos = q0 + m;
        bool ok = (key <= qpos) && (qpos - key < WIN);
        sc[j][i] = ok ? s[i] * scale : -3.0e38f;
      }
    }
    float corr[8], lsum[8];
#pragma unroll
    for (int i = 0; i < 8; ++i) {
      float v = fmaxf(sc[0][i], sc[1][i]);
#pragma unroll
      for (int xm = 1; xm < 16; xm <<= 1) v = fmaxf(v, __shfl_xor(v, xm, 32));
      float nm = fmaxf(rmax[i], v);
      corr[i] = __expf(rmax[i] - nm);
      rmax[i] = nm;
      lsum[i] = 0.0f;
    }
#pragma unroll
    for (int j = 0; j < 2; ++j) {
#pragma unroll
      for (int i = 0; i < 8; ++i) {
        int m = i + 8 * khalf;
        float p = (sc[j][i] < -1.0e37f) ? 0.0f : __expf(sc[j][i] - rmax[i]);
        lsum[i] += p;
        amy[m * 40 + 16 * j + r] = (_Float16)p;
      }
    }
#pragma unroll
    for (int i = 0; i < 8; ++i) {
      float v = lsum[i];
#pragma unroll
      for (int xm = 1; xm < 16; xm <<= 1) v += __shfl_xor(v, xm, 32);
      rsum[i] = rsum[i] * corr[i] + v;
      acc0[i] *= corr[i];
      acc1[i] *= corr[i];
      acc2[i] *= corr[i];
      acc3[i] *= corr[i];
    }
    // Order the per-wave LDS stores before the A-fragment reload.
    asm volatile("s_wait_dscnt 0" ::: "memory");
    v16h af = ldA(amy + r * 40 + khalf * 16);
    v16h vf0 = ldB(vbase + (size_t)(0 + r) * SS + kc + khalf * 16);
    v16h vf1 = ldB(vbase + (size_t)(16 + r) * SS + kc + khalf * 16);
    v16h vf2 = ldB(vbase + (size_t)(32 + r) * SS + kc + khalf * 16);
    v16h vf3 = ldB(vbase + (size_t)(48 + r) * SS + kc + khalf * 16);
    acc0 = wmma16(af, vf0, acc0);
    acc1 = wmma16(af, vf1, acc1);
    acc2 = wmma16(af, vf2, acc2);
    acc3 = wmma16(af, vf3, acc3);
  }

  _Float16* yb = y + ((size_t)(b * NH + h)) * SS * HDIM;
#pragma unroll
  for (int i = 0; i < 8; ++i) {
    int m = i + 8 * khalf;
    float inv = 1.0f / rsum[i];
    yb[(size_t)(q0 + m) * HDIM + r] = (_Float16)(acc0[i] * inv);
    yb[(size_t)(q0 + m) * HDIM + 16 + r] = (_Float16)(acc1[i] * inv);
    yb[(size_t)(q0 + m) * HDIM + 32 + r] = (_Float16)(acc2[i] * inv);
    yb[(size_t)(q0 + m) * HDIM + 48 + r] = (_Float16)(acc3[i] * inv);
  }
}

__global__ void pair_mix_kernel(const _Float16* __restrict__ y,
                                const float* __restrict__ pm,
                                _Float16* __restrict__ ym) {
  int n = BB * SS * DMODEL;
  for (int o = blockIdx.x * blockDim.x + threadIdx.x; o < n;
       o += gridDim.x * blockDim.x) {
    int d = o & 63;
    int h = (o >> 6) & 15;
    int sb = o >> 10;  // b*S + s
    int s = sb & (SS - 1);
    int b = sb >> 11;
    int p = h >> 1, oi = h & 1;
    float y0 = (float)y[(((size_t)(b * NH + 2 * p)) * SS + s) * HDIM + d];
    float y1 = (float)y[(((size_t)(b * NH + 2 * p + 1)) * SS + s) * HDIM + d];
    ym[o] = (_Float16)(pm[p * 4 + oi * 2 + 0] * y0 +
                       pm[p * 4 + oi * 2 + 1] * y1);
  }
}

extern "C" void kernel_launch(void* const* d_in, const int* in_sizes, int n_in,
                              void* d_out, int out_size, void* d_ws,
                              size_t ws_size, hipStream_t stream) {
  const float* x = (const float*)d_in[0];
  const float* Wq = (const float*)d_in[1];
  const float* Wk = (const float*)d_in[2];
  const float* Wv = (const float*)d_in[3];
  const float* Wo = (const float*)d_in[4];
  const float* qg = (const float*)d_in[5];
  const float* pm = (const float*)d_in[6];

  char* ws = (char*)d_ws;
  const int NT = BB * SS;  // 4096 tokens
  _Float16* wq_h = (_Float16*)(ws + 0);         // 1024x1024
  _Float16* wk_h = (_Float16*)(ws + 2097152);   // 256x1024
  _Float16* wv_h = (_Float16*)(ws + 2621440);   // 256x1024
  _Float16* wo_h = (_Float16*)(ws + 3145728);   // 1024x1024
  _Float16* k_h  = (_Float16*)(ws + 5242880);   // [b,kvh,s,hd]
  _Float16* vt_h = (_Float16*)(ws + 7340032);   // [b,kvh,hd,s]
  _Float16* klin = (_Float16*)(ws + 9437184);   // 4096x256
  _Float16* vlin = (_Float16*)(ws + 11534336);  // 4096x256
  _Float16* x_h  = (_Float16*)(ws + 13631488);  // 4096x1024 (later reused as y)
  _Float16* qlin = (_Float16*)(ws + 22020096);  // 4096x1024 (later reused as ym)
  _Float16* q_h  = (_Float16*)(ws + 30408704);  // [b,h,s,hd]
  _Float16* y_h  = x_h;   // x consumed by the QKV GEMMs before attention
  _Float16* ym_h = qlin;  // q_lin consumed by rms_rope before pair_mix

  cvt_f16_kernel<<<4096, 256, 0, stream>>>(x, x_h, NT * DMODEL);
  cvt_f16_kernel<<<2048, 256, 0, stream>>>(Wq, wq_h, DMODEL * DMODEL);
  cvt_f16_kernel<<<512, 256, 0, stream>>>(Wk, wk_h, NKV * HDIM * DMODEL);
  cvt_f16_kernel<<<512, 256, 0, stream>>>(Wv, wv_h, NKV * HDIM * DMODEL);
  cvt_f16_kernel<<<2048, 256, 0, stream>>>(Wo, wo_h, DMODEL * DMODEL);

  gemm_nt_wmma<<<dim3(DMODEL / 64, NT / 128), 256, 0, stream>>>(
      x_h, wq_h, nullptr, qlin, NT, DMODEL, DMODEL);
  gemm_nt_wmma<<<dim3(256 / 64, NT / 128), 256, 0, stream>>>(
      x_h, wk_h, nullptr, klin, NT, 256, DMODEL);
  gemm_nt_wmma<<<dim3(256 / 64, NT / 128), 256, 0, stream>>>(
      x_h, wv_h, nullptr, vlin, NT, 256, DMODEL);

  rms_rope_kernel<<<(BB * NH * SS * 32) / 256, 256, 0, stream>>>(qlin, q_h, qg, NH);
  rms_rope_kernel<<<(BB * NKV * SS * 32) / 256, 256, 0, stream>>>(klin, k_h, nullptr, NKV);
  v_transpose_kernel<<<4096, 256, 0, stream>>>(vlin, vt_h);

  attn_kernel<<<(BB * NH * (SS / 16)) / 4, 128, 0, stream>>>(q_h, k_h, vt_h, y_h);

  pair_mix_kernel<<<8192, 256, 0, stream>>>(y_h, pm, ym_h);

  gemm_nt_wmma<<<dim3(DMODEL / 64, NT / 128), 256, 0, stream>>>(
      ym_h, wo_h, (float*)d_out, nullptr, NT, DMODEL, DMODEL);
}
